// HeterogeneousGAT_46291157516720
// MI455X (gfx1250) — matrix operand order repaired and checked by reading, not compiled, not run on hardware
//
#include <hip/hip_runtime.h>
#include <hip/hip_bf16.h>
#include <math.h>

// ---------------------------------------------------------------------------
// HeterogeneousGAT forward for MI455X (gfx1250, wave32, WMMA).
// MLP GEMMs: v_wmma_f32_16x16x32_f16, weights staged per-block in LDS as
// transposed f16 with XOR bank swizzle (B fragments = 2x ds_load_b128, no
// converts in the inner loop). Activation selection is compile-time per
// instantiation. Edge gather/scatter: global f32 atomics.
// ---------------------------------------------------------------------------

typedef __attribute__((ext_vector_type(16))) _Float16 v16h;
typedef __attribute__((ext_vector_type(8)))  _Float16 v8h;
typedef __attribute__((ext_vector_type(8)))  float    v8f;

static __device__ __forceinline__ float eluf(float x)   { return x > 0.f ? x : expm1f(x); }
static __device__ __forceinline__ float lrelu2(float x) { return x > 0.f ? x : 0.2f * x; }

// XOR-swizzled element offset into a row-major [rows x KDIM] f16 LDS tile.
// 8-half (16B) groups are permuted by row so 16 lanes reading the same k-group
// across consecutive rows hit distinct bank groups; 16B alignment preserved.
template <int KDIM>
static __device__ __forceinline__ int swz(int row, int k) {
  constexpr int GM = KDIM / 8 - 1;
  return row * KDIM + ((((k >> 3) ^ row) & GM) << 3) + (k & 7);
}

// ---------------------------- utility kernels ------------------------------

__global__ void zero_kernel(float* __restrict__ p, long n) {
  long i = (long)blockIdx.x * blockDim.x + threadIdx.x;
  long stride = (long)gridDim.x * blockDim.x;
  for (; i < n; i += stride) p[i] = 0.f;
}

// Y[i,0:8] = X[i,0:din] @ W(din x 8) + b ; optionally self_att[i]=lrelu(sum r*r)
__global__ void lin8_kernel(const float* __restrict__ X, int din,
                            const float* __restrict__ W, const float* __restrict__ b,
                            float* __restrict__ Y, float* __restrict__ selfatt, int n) {
  int i = blockIdx.x * blockDim.x + threadIdx.x;
  if (i >= n) return;
  float ss = 0.f;
  #pragma unroll
  for (int k = 0; k < 8; ++k) {
    float s = b[k];
    for (int j = 0; j < din; ++j) s += X[(long)i * din + j] * W[j * 8 + k];
    Y[(long)i * 8 + k] = s;
    ss += s * s;
  }
  if (selfatt) selfatt[i] = lrelu2(ss);
}

// per-edge attention + scatter into cross[R][3][8]
__global__ void edge_att_kernel(const int* __restrict__ src, const int* __restrict__ dst,
                                const float* __restrict__ o_lin, const float* __restrict__ r_lin,
                                const float* __restrict__ att_ro, float* __restrict__ cross, int E) {
  long i = (long)blockIdx.x * blockDim.x + threadIdx.x;
  long stride = (long)gridDim.x * blockDim.x;
  for (; i < E; i += stride) {
    const int s = src[i], d = dst[i];
    float xj[8], vi[8];
    #pragma unroll
    for (int k = 0; k < 8; ++k) xj[k] = o_lin[(long)s * 8 + k];
    #pragma unroll
    for (int k = 0; k < 8; ++k) vi[k] = r_lin[(long)d * 8 + k];
    #pragma unroll
    for (int h = 0; h < 3; ++h) {
      float t = 0.f;
      #pragma unroll
      for (int k = 0; k < 8; ++k) t += vi[k] * att_ro[h * 16 + k] + xj[k] * att_ro[h * 16 + 8 + k];
      t = lrelu2(t);
      float* cp = cross + ((long)d * 3 + h) * 8;
      #pragma unroll
      for (int k = 0; k < 8; ++k) atomicAdd(cp + k, t * xj[k]);
    }
  }
}

// softmax over [self_att, cross[:,h,:]] per head, combine, elu
__global__ void res_fin_kernel(const float* __restrict__ r_lin, const float* __restrict__ selfa,
                               const float* __restrict__ cross, float* __restrict__ resOut, int n) {
  int i = blockIdx.x * blockDim.x + threadIdx.x;
  if (i >= n) return;
  const float sa = selfa[i];
  float acc[8];
  #pragma unroll
  for (int k = 0; k < 8; ++k) acc[k] = 0.f;
  float ssum = 0.f;
  #pragma unroll
  for (int h = 0; h < 3; ++h) {
    const float* cr = cross + ((long)i * 3 + h) * 8;
    float cv[8], m = sa;
    #pragma unroll
    for (int k = 0; k < 8; ++k) { cv[k] = cr[k]; m = fmaxf(m, cv[k]); }
    float e0 = expf(sa - m), den = e0;
    float ek[8];
    #pragma unroll
    for (int k = 0; k < 8; ++k) { ek[k] = expf(cv[k] - m); den += ek[k]; }
    float inv = 1.f / den;
    ssum += e0 * inv;
    #pragma unroll
    for (int k = 0; k < 8; ++k) acc[k] += ek[k] * inv * cv[k];
  }
  #pragma unroll
  for (int k = 0; k < 8; ++k)
    resOut[(long)i * 8 + k] = eluf(ssum * r_lin[(long)i * 8 + k] + acc[k]);
}

// precedence-edge segment sums (+ in/out degree counts)
__global__ void prec_kernel(const int* __restrict__ src, const int* __restrict__ dst,
                            const float* __restrict__ ops, int din,
                            float* __restrict__ psum, float* __restrict__ ssum,
                            float* __restrict__ cin, float* __restrict__ cou, int E) {
  long i = (long)blockIdx.x * blockDim.x + threadIdx.x;
  long stride = (long)gridDim.x * blockDim.x;
  for (; i < E; i += stride) {
    const int s = src[i], d = dst[i];
    atomicAdd(&cin[d], 1.f);
    atomicAdd(&cou[s], 1.f);
    for (int k = 0; k < din; ++k) {
      atomicAdd(&psum[(long)d * 8 + k], ops[(long)s * din + k]);
      atomicAdd(&ssum[(long)s * 8 + k], ops[(long)d * din + k]);
    }
  }
}

__global__ void seg_mean_kernel(float* __restrict__ sum, const float* __restrict__ cnt,
                                int din, int n) {
  long i = (long)blockIdx.x * blockDim.x + threadIdx.x;
  if (i >= (long)n * 8) return;
  int k = (int)(i & 7);
  int r = (int)(i >> 3);
  if (k < din) sum[i] /= fmaxf(cnt[r], 1.f);
}

// requirement-edge gather of resource features into per-op sums (+ counts)
__global__ void req_agg_kernel(const int* __restrict__ src, const int* __restrict__ dst,
                               const float* __restrict__ res, float* __restrict__ agg,
                               float* __restrict__ cnt, int E) {
  long i = (long)blockIdx.x * blockDim.x + threadIdx.x;
  long stride = (long)gridDim.x * blockDim.x;
  for (; i < E; i += stride) {
    const int s = src[i], d = dst[i];
    atomicAdd(&cnt[s], 1.f);
    #pragma unroll
    for (int k = 0; k < 8; ++k) atomicAdd(&agg[(long)s * 8 + k], res[(long)d * 8 + k]);
  }
}

// column sums of an n x 8 matrix -> atomicAdd into out[0..7]
__launch_bounds__(256)
__global__ void colsum8_kernel(const float* __restrict__ X, int n, float* __restrict__ out) {
  float acc[8];
  #pragma unroll
  for (int k = 0; k < 8; ++k) acc[k] = 0.f;
  for (long i = (long)blockIdx.x * blockDim.x + threadIdx.x; i < n;
       i += (long)gridDim.x * blockDim.x) {
    #pragma unroll
    for (int k = 0; k < 8; ++k) acc[k] += X[i * 8 + k];
  }
  __shared__ float red[8][256];
  #pragma unroll
  for (int k = 0; k < 8; ++k) red[k][threadIdx.x] = acc[k];
  __syncthreads();
  for (int off = 128; off > 0; off >>= 1) {
    if ((int)threadIdx.x < off) {
      #pragma unroll
      for (int k = 0; k < 8; ++k) red[k][threadIdx.x] += red[k][threadIdx.x + off];
    }
    __syncthreads();
  }
  if (threadIdx.x < 8) atomicAdd(&out[threadIdx.x], red[threadIdx.x][0]);
}

// actor_in = [ops(8) | agg/cnt(8) | g(16)]
__global__ void actor_in_kernel(const float* __restrict__ ops, const float* __restrict__ agg,
                                const float* __restrict__ cnt, const float* __restrict__ gsum,
                                float nops, float nres, float* __restrict__ out, int n) {
  int i = blockIdx.x * blockDim.x + threadIdx.x;
  if (i >= n) return;
  float inv = 1.f / fmaxf(cnt[i], 1.f);
  #pragma unroll
  for (int k = 0; k < 8; ++k) {
    out[(long)i * 32 + k]      = ops[(long)i * 8 + k];
    out[(long)i * 32 + 8 + k]  = agg[(long)i * 8 + k] * inv;
    out[(long)i * 32 + 16 + k] = gsum[k] / nops;
    out[(long)i * 32 + 24 + k] = gsum[8 + k] / nres;
  }
}

// critic: single row 16 -> 64 -> 64 -> 1 (tanh), scalar is fine
__global__ void critic_kernel(const float* __restrict__ gsum, float nops, float nres,
                              const float* __restrict__ W0, const float* __restrict__ b0,
                              const float* __restrict__ W1, const float* __restrict__ b1,
                              const float* __restrict__ W2, const float* __restrict__ b2,
                              float* __restrict__ out) {
  if (threadIdx.x != 0 || blockIdx.x != 0) return;
  float g[16];
  for (int k = 0; k < 8; ++k) { g[k] = gsum[k] / nops; g[8 + k] = gsum[8 + k] / nres; }
  float h1[64];
  for (int j = 0; j < 64; ++j) {
    float s = b0[j];
    for (int k = 0; k < 16; ++k) s += g[k] * W0[k * 64 + j];
    h1[j] = tanhf(s);
  }
  float h2[64];
  for (int j = 0; j < 64; ++j) {
    float s = b1[j];
    for (int k = 0; k < 64; ++k) s += h1[k] * W1[k * 64 + j];
    h2[j] = tanhf(s);
  }
  float v = b2[0];
  for (int j = 0; j < 64; ++j) v += h2[j] * W2[j];
  out[0] = v;
}

// ----------------------- WMMA 3-layer MLP kernel ---------------------------
// din (<=32) -> HID -> HID -> dout (<=16). 4 waves/block, one 16-row tile per
// wave per iteration (wave-level grid-stride). Weights staged once per block
// in LDS (transposed, f16, XOR-swizzled) -> B fragments are 2x ds_load_b128.
// Hidden activations staged per-wave in swizzled LDS (C-layout -> A-layout).
// ACT_TANH / PRE_ELU are compile-time so each instantiation carries exactly
// one activation path (no per-element runtime branches around the WMMAs).
// Fragment layouts: A 16x32 f16: lane m=lane&15, half h -> K = 8*(lane>>4) +
// (h<8 ? h : h+8). B 32x16: lane n=lane&15, half h -> same K as A.
// C/D: VGPR v -> row v + 8*(lane>>4), col lane&15.
template <int HID, int ACT_TANH, int PRE_ELU>
__launch_bounds__(128)
__global__ void mlp3_wmma(const float* __restrict__ X, int ldx, int din,
                          const float* __restrict__ W0, const float* __restrict__ b0,
                          const float* __restrict__ W1, const float* __restrict__ b1,
                          const float* __restrict__ W2, const float* __restrict__ b2,
                          int dout,
                          float* __restrict__ Y, int ldy, int ycol,
                          const float* __restrict__ gate, int interior, int nrows) {
  constexpr int NT = HID / 16;  // hidden column tiles
  constexpr int NK = HID / 32;  // K-steps over hidden dim
  __shared__ __align__(16) _Float16 sW0[32 * HID];     // W0^T, K padded to 32
  __shared__ __align__(16) _Float16 sW1[HID * HID];    // W1^T
  __shared__ __align__(16) _Float16 sW2[16 * HID];     // W2^T, N padded to 16
  __shared__ __align__(16) _Float16 sH[4][16 * HID];   // per-wave activations
  __shared__ float sB0[HID], sB1[HID], sB2[16];

  const int tid = threadIdx.x;
  // ---- stage weights/biases once per block (coalesced global reads) ----
  for (int idx = tid; idx < 32 * HID; idx += 128) {
    const int k = idx / HID, n = idx % HID;
    sW0[swz<32>(n, k)] = (_Float16)(k < din ? W0[(long)k * HID + n] : 0.f);
  }
  for (int idx = tid; idx < HID * HID; idx += 128) {
    const int k = idx / HID, n = idx % HID;
    sW1[swz<HID>(n, k)] = (_Float16)W1[(long)k * HID + n];
  }
  for (int idx = tid; idx < 16 * HID; idx += 128) {
    const int k = idx >> 4, n = idx & 15;
    sW2[swz<HID>(n, k)] = (_Float16)(n < dout ? W2[(long)k * dout + n] : 0.f);
  }
  for (int idx = tid; idx < HID; idx += 128) { sB0[idx] = b0[idx]; sB1[idx] = b1[idx]; }
  if (tid < 16) sB2[tid] = (tid < dout) ? b2[tid] : 0.f;
  __syncthreads();

  const int wave = tid >> 5;
  const int lane = tid & 31;
  const int half = lane >> 4;
  const int mcol = lane & 15;
  _Float16* H = sH[wave];

  auto act = [](float x) -> float {
    if constexpr (ACT_TANH) return tanhf(x); else return eluf(x);
  };

  const int ntiles = (nrows + 15) >> 4;
  const int wstride = gridDim.x * 4;
  for (int tile = blockIdx.x * 4 + wave; tile < ntiles; tile += wstride) {
    const int row0 = tile << 4;

    // ---- A0: 16x32 input tile (zero-padded past din) ----
    v16h a0;
    {
      const int r = row0 + mcol;
      const bool rok = r < nrows;
      const long base = (long)r * ldx;
      const int kb = half * 8;
      #pragma unroll
      for (int h = 0; h < 16; ++h) {
        const int k = kb + (h < 8 ? h : h + 8);
        float v = 0.f;
        if (rok && k < din) {
          v = X[base + k];
          if constexpr (PRE_ELU) v = eluf(v);
        }
        a0[h] = (_Float16)v;
      }
    }

    // ---- layer 1: H1 = act(A0 @ W0 + b0), single K=32 step ----
    #pragma unroll
    for (int nt = 0; nt < NT; ++nt) {
      const int n = nt * 16 + mcol;
      const float bv = sB0[n];
      v8f c = {bv, bv, bv, bv, bv, bv, bv, bv};
      const int kb = half * 8;
      const v8h blo = *(const v8h*)&sW0[swz<32>(n, kb)];
      const v8h bhi = *(const v8h*)&sW0[swz<32>(n, kb + 16)];
      v16h b;
      #pragma unroll
      for (int h = 0; h < 8; ++h) { b[h] = blo[h]; b[h + 8] = bhi[h]; }
      c = __builtin_amdgcn_wmma_f32_16x16x32_f16(false, a0, false, b, (short)0, c, false, false);
      #pragma unroll
      for (int v = 0; v < 8; ++v)
        H[swz<HID>(v + half * 8, n)] = (_Float16)act(c[v]);
    }

    // ---- layer 2: H2 = act(H1 @ W1 + b1) ----
    v8f acc2[NT];
    #pragma unroll
    for (int nt = 0; nt < NT; ++nt) {
      const int n = nt * 16 + mcol;
      const float bv = sB1[n];
      v8f c = {bv, bv, bv, bv, bv, bv, bv, bv};
      #pragma unroll
      for (int ks = 0; ks < NK; ++ks) {
        const int k0 = ks * 32 + half * 8;
        const v8h alo = *(const v8h*)&H[swz<HID>(mcol, k0)];
        const v8h ahi = *(const v8h*)&H[swz<HID>(mcol, k0 + 16)];
        v16h a;
        #pragma unroll
        for (int h = 0; h < 8; ++h) { a[h] = alo[h]; a[h + 8] = ahi[h]; }
        const int kb = ks * 32 + half * 8;
        const v8h blo = *(const v8h*)&sW1[swz<HID>(n, kb)];
        const v8h bhi = *(const v8h*)&sW1[swz<HID>(n, kb + 16)];
        v16h b;
        #pragma unroll
        for (int h = 0; h < 8; ++h) { b[h] = blo[h]; b[h + 8] = bhi[h]; }
        c = __builtin_amdgcn_wmma_f32_16x16x32_f16(false, a, false, b, (short)0, c, false, false);
      }
      #pragma unroll
      for (int v = 0; v < 8; ++v) acc2[nt][v] = act(c[v]);
    }
    // overwrite H with H2 (all H1 reads done; same-wave LDS ops stay in order)
    #pragma unroll
    for (int nt = 0; nt < NT; ++nt) {
      const int n = nt * 16 + mcol;
      #pragma unroll
      for (int v = 0; v < 8; ++v) H[swz<HID>(v + half * 8, n)] = (_Float16)acc2[nt][v];
    }

    // ---- layer 3: Y = H2 @ W2 + b2 (no activation) ----
    {
      const int n = mcol;
      const float bv = sB2[n];
      v8f c = {bv, bv, bv, bv, bv, bv, bv, bv};
      #pragma unroll
      for (int ks = 0; ks < NK; ++ks) {
        const int k0 = ks * 32 + half * 8;
        const v8h alo = *(const v8h*)&H[swz<HID>(mcol, k0)];
        const v8h ahi = *(const v8h*)&H[swz<HID>(mcol, k0 + 16)];
        v16h a;
        #pragma unroll
        for (int h = 0; h < 8; ++h) { a[h] = alo[h]; a[h + 8] = ahi[h]; }
        const int kb = ks * 32 + half * 8;
        const v8h blo = *(const v8h*)&sW2[swz<HID>(n, kb)];
        const v8h bhi = *(const v8h*)&sW2[swz<HID>(n, kb + 16)];
        v16h b;
        #pragma unroll
        for (int h = 0; h < 8; ++h) { b[h] = blo[h]; b[h + 8] = bhi[h]; }
        c = __builtin_amdgcn_wmma_f32_16x16x32_f16(false, a, false, b, (short)0, c, false, false);
      }
      if (n < dout) {
        #pragma unroll
        for (int v = 0; v < 8; ++v) {
          const int r = row0 + v + half * 8;
          if (r < nrows) {
            float x = c[v];
            if (gate && gate[r] <= 0.f) x = 0.f;
            if (interior && (r == 0 || r == nrows - 1)) x = 0.f;
            Y[(long)r * ldy + ycol + n] = x;
          }
        }
      }
    }
  }
}

// ------------------------------ host driver --------------------------------

extern "C" void kernel_launch(void* const* d_in, const int* in_sizes, int n_in,
                              void* d_out, int out_size, void* d_ws, size_t ws_size,
                              hipStream_t stream) {
  (void)n_in; (void)ws_size; (void)out_size;
  const float* operations = (const float*)d_in[0];   // N x 6
  const float* resources  = (const float*)d_in[1];   // R x 3
  const int* prec_src = (const int*)d_in[2];
  const int* prec_dst = (const int*)d_in[3];
  const int* req_src  = (const int*)d_in[4];
  const int* req_dst  = (const int*)d_in[5];
  const int N  = in_sizes[0] / 6;
  const int R  = in_sizes[1] / 3;
  const int EP = in_sizes[2];
  const int EQ = in_sizes[4];

  // params pytree leaves (dict keys sorted): actor, critic, op[0..1], res[0..1]
  int p = 6;
  auto F = [&](int i) { return (const float*)d_in[i]; };
  const float* actW0 = F(p++); const float* actb0 = F(p++);
  const float* actW1 = F(p++); const float* actb1 = F(p++);
  const float* actW2 = F(p++); const float* actb2 = F(p++);
  const float* crW0 = F(p++); const float* crb0 = F(p++);
  const float* crW1 = F(p++); const float* crb1 = F(p++);
  const float* crW2 = F(p++); const float* crb2 = F(p++);
  // op layers: MLPs in key order comb, pred, res, same, succ; each (W,b) x 3
  enum { M_COMB = 0, M_PRED = 1, M_RES = 2, M_SAME = 3, M_SUCC = 4 };
  const float* opW[2][5][3]; const float* opB[2][5][3];
  for (int l = 0; l < 2; ++l)
    for (int m = 0; m < 5; ++m)
      for (int j = 0; j < 3; ++j) { opW[l][m][j] = F(p++); opB[l][m][j] = F(p++); }
  // res layers: att_ro, ot.W, ot.b, rt.W, rt.b
  const float* attro[2]; const float* otW[2]; const float* otB[2];
  const float* rtW[2];   const float* rtB[2];
  for (int l = 0; l < 2; ++l) {
    attro[l] = F(p++); otW[l] = F(p++); otB[l] = F(p++); rtW[l] = F(p++); rtB[l] = F(p++);
  }

  // workspace carve (floats)
  float* w = (float*)d_ws;
  auto carve = [&](long n) { float* r = w; w += n; return r; };
  float* o_lin = carve((long)N * 8);
  float* r_lin = carve((long)R * 8);
  float* selfa = carve(R);
  float* cross = carve((long)R * 24);
  float* res1  = carve((long)R * 8);
  float* res2  = carve((long)R * 8);
  float* psum  = carve((long)N * 8);
  float* ssum  = carve((long)N * 8);
  float* agg   = carve((long)N * 8);
  float* cin   = carve(N);
  float* cou   = carve(N);
  float* creq  = carve(N);
  float* ops1  = carve((long)N * 8);
  float* ops2  = carve((long)N * 8);
  float* gsum  = carve(16);
  float* hbuf  = carve((long)N * 32);
  float* actin = hbuf;  // h retired before actor input is built

  auto zero = [&](float* ptr, long n) {
    int blocks = (int)((n + 255) / 256); if (blocks > 8192) blocks = 8192;
    zero_kernel<<<blocks, 256, 0, stream>>>(ptr, n);
  };
  const int tiles = (N + 15) / 16;
  int mlpBlocks = (tiles + 3) / 4;
  if (mlpBlocks > 512) mlpBlocks = 512;  // grid-stride; amortizes LDS staging

  const float* opsCur = operations; int dinO = 6;
  const float* resCur = resources;  int dinR = 3;
  for (int l = 0; l < 2; ++l) {
    float* resOut = (l == 0) ? res1 : res2;
    float* opsOut = (l == 0) ? ops1 : ops2;

    // ---- res layer ----
    lin8_kernel<<<(R + 255) / 256, 256, 0, stream>>>(resCur, dinR, rtW[l], rtB[l], r_lin, selfa, R);
    lin8_kernel<<<(N + 255) / 256, 256, 0, stream>>>(opsCur, dinO, otW[l], otB[l], o_lin, nullptr, N);
    zero(cross, (long)R * 24);
    edge_att_kernel<<<4096, 256, 0, stream>>>(req_src, req_dst, o_lin, r_lin, attro[l], cross, EQ);
    res_fin_kernel<<<(R + 255) / 256, 256, 0, stream>>>(r_lin, selfa, cross, resOut, R);

    // ---- op layer ----
    zero(psum, (long)N * 8); zero(ssum, (long)N * 8); zero(agg, (long)N * 8);
    zero(cin, N); zero(cou, N); zero(creq, N);
    prec_kernel<<<2048, 256, 0, stream>>>(prec_src, prec_dst, opsCur, dinO, psum, ssum, cin, cou, EP);
    seg_mean_kernel<<<((long)N * 8 + 255) / 256, 256, 0, stream>>>(psum, cin, dinO, N);
    seg_mean_kernel<<<((long)N * 8 + 255) / 256, 256, 0, stream>>>(ssum, cou, dinO, N);
    req_agg_kernel<<<4096, 256, 0, stream>>>(req_src, req_dst, resOut, agg, creq, EQ);

    mlp3_wmma<128, 0, 0><<<mlpBlocks, 128, 0, stream>>>(psum, 8, dinO,
        opW[l][M_PRED][0], opB[l][M_PRED][0], opW[l][M_PRED][1], opB[l][M_PRED][1],
        opW[l][M_PRED][2], opB[l][M_PRED][2], 8, hbuf, 32, 0, cin, 0, N);
    mlp3_wmma<128, 0, 0><<<mlpBlocks, 128, 0, stream>>>(ssum, 8, dinO,
        opW[l][M_SUCC][0], opB[l][M_SUCC][0], opW[l][M_SUCC][1], opB[l][M_SUCC][1],
        opW[l][M_SUCC][2], opB[l][M_SUCC][2], 8, hbuf, 32, 8, cou, 0, N);
    mlp3_wmma<128, 0, 0><<<mlpBlocks, 128, 0, stream>>>(agg, 8, 8,
        opW[l][M_RES][0], opB[l][M_RES][0], opW[l][M_RES][1], opB[l][M_RES][1],
        opW[l][M_RES][2], opB[l][M_RES][2], 8, hbuf, 32, 16, nullptr, 0, N);
    mlp3_wmma<128, 0, 0><<<mlpBlocks, 128, 0, stream>>>(opsCur, dinO, dinO,
        opW[l][M_SAME][0], opB[l][M_SAME][0], opW[l][M_SAME][1], opB[l][M_SAME][1],
        opW[l][M_SAME][2], opB[l][M_SAME][2], 8, hbuf, 32, 24, nullptr, 0, N);
    mlp3_wmma<128, 0, 1><<<mlpBlocks, 128, 0, stream>>>(hbuf, 32, 32,
        opW[l][M_COMB][0], opB[l][M_COMB][0], opW[l][M_COMB][1], opB[l][M_COMB][1],
        opW[l][M_COMB][2], opB[l][M_COMB][2], 8, opsOut, 8, 0, nullptr, /*interior=*/1, N);

    opsCur = opsOut; dinO = 8;
    resCur = resOut; dinR = 8;
  }

  // ---- global pooling + heads ----
  zero(gsum, 16);
  colsum8_kernel<<<256, 256, 0, stream>>>(ops2, N, gsum);
  colsum8_kernel<<<8, 256, 0, stream>>>(res2, R, gsum + 8);
  actor_in_kernel<<<(N + 255) / 256, 256, 0, stream>>>(ops2, agg, creq, gsum,
                                                       (float)N, (float)R, actin, N);
  mlp3_wmma<64, 1, 0><<<mlpBlocks, 128, 0, stream>>>(actin, 32, 32,
      actW0, actb0, actW1, actb1, actW2, actb2, 1,
      (float*)d_out, 1, 0, nullptr, 0, N);
  critic_kernel<<<1, 1, 0, stream>>>(gsum, (float)N, (float)R,
      crW0, crb0, crW1, crb1, crW2, crb2, (float*)d_out + N);
}
